// MultiHeadAttentionQuantum_65481071408433
// MI455X (gfx1250) — compile-verified, matrix operand order checked by reading, not run on hardware
//
#include <hip/hip_runtime.h>
#include <hip/hip_bf16.h>
#include <math.h>

#define EMBED     1024
#define HEADS     16
#define HEAD_DIM  64
#define BATCH     8
#define SEQ       4096
#define MROWS     (BATCH * SEQ)   // 32768 tokens

typedef _Float16 half8  __attribute__((ext_vector_type(8)));
typedef _Float16 half16 __attribute__((ext_vector_type(16)));
typedef float    float8 __attribute__((ext_vector_type(8)));

// ---------------------------------------------------------------------------
// f32 -> f16 conversion (grid-stride)
// ---------------------------------------------------------------------------
__global__ void cvt_f32_to_f16(const float* __restrict__ in,
                               _Float16* __restrict__ out, int n) {
    int i = blockIdx.x * blockDim.x + threadIdx.x;
    int stride = gridDim.x * blockDim.x;
    for (; i < n; i += stride) out[i] = (_Float16)in[i];
}

// ---------------------------------------------------------------------------
// Fragment load for V_WMMA_F32_16X16X32_F16.
// 16-bit A layout (ISA 7.12.2): lane l holds row M = l&15; VGPR0-3 hold K =
// base+0..7 (base = (l>=16)?8:0), VGPR4-7 hold K = base+16..23, i.e. two
// contiguous 8-half (16-byte) runs at p[0..7] and p[16..23] where p already
// includes the per-lane base offset. B (=W^T) mirrors this with lane = output
// column (= row of W), so both A and B runs are contiguous in memory.
// ---------------------------------------------------------------------------
__device__ __forceinline__ half16 ldfrag(const _Float16* p) {
    half8 lo = *(const half8*)(p);
    half8 hi = *(const half8*)(p + 16);
    return __builtin_shufflevector(lo, hi,
        0, 1, 2, 3, 4, 5, 6, 7, 8, 9, 10, 11, 12, 13, 14, 15);
}

// ---------------------------------------------------------------------------
// y = A @ W^T  (A: [M x 1024] f16 row-major, W: [1024 x 1024] f16 row-major)
// Each wave computes a 32(M) x 64(N) tile: 2 A-fragments + 4 B-fragments per
// K-step feed 8 WMMAs (21 FLOP/B from cache vs 13 for a 16x64 tile).
// Block = (32,4): 4 waves tile N -> 32 x 256 per block.
// Grid = (M/32, 1024/256).
// mode 0: out = cos(acc + theta[n % 64]) stored f16   (quantum epilogue)
// mode 1: out = acc stored f32 (final projection to d_out)
// ---------------------------------------------------------------------------
__global__ __launch_bounds__(128)
void gemm_wmma(const _Float16* __restrict__ A,
               const _Float16* __restrict__ W,
               _Float16* __restrict__ outH,
               float* __restrict__ outF,
               const float* __restrict__ theta,
               int mode) {
    const int lane  = threadIdx.x;           // 0..31 (wave32)
    const int wave  = threadIdx.y;           // 0..3
    const int tm    = blockIdx.x * 32;       // M tile base (2 x 16 rows)
    const int nbase = (blockIdx.y * 4 + wave) * 64;
    const int khi   = (lane >> 4) * 8;       // per-lane K sub-offset
    const int ln    = lane & 15;

    const _Float16* ap0 = A + (size_t)(tm +  0 + ln) * EMBED + khi;
    const _Float16* ap1 = A + (size_t)(tm + 16 + ln) * EMBED + khi;
    const _Float16* bp0 = W + (size_t)(nbase +  0 + ln) * EMBED + khi;
    const _Float16* bp1 = W + (size_t)(nbase + 16 + ln) * EMBED + khi;
    const _Float16* bp2 = W + (size_t)(nbase + 32 + ln) * EMBED + khi;
    const _Float16* bp3 = W + (size_t)(nbase + 48 + ln) * EMBED + khi;

    float8 acc[2][4] = {};

    for (int kk = 0; kk < EMBED; kk += 32) {
        // prefetch the streaming A operand one K-block ahead
        // (lowers to global_prefetch_b8; harmlessly speculative at row end)
        __builtin_prefetch(ap0 + kk + 32, 0, 1);
        __builtin_prefetch(ap1 + kk + 32, 0, 1);

        half16 a0 = ldfrag(ap0 + kk);
        half16 a1 = ldfrag(ap1 + kk);

        half16 b0 = ldfrag(bp0 + kk);
        acc[0][0] = __builtin_amdgcn_wmma_f32_16x16x32_f16(false, a0, false, b0,
                                                 (short)0, acc[0][0], false, false);
        acc[1][0] = __builtin_amdgcn_wmma_f32_16x16x32_f16(false, a1, false, b0,
                                                 (short)0, acc[1][0], false, false);
        half16 b1 = ldfrag(bp1 + kk);
        acc[0][1] = __builtin_amdgcn_wmma_f32_16x16x32_f16(false, a0, false, b1,
                                                 (short)0, acc[0][1], false, false);
        acc[1][1] = __builtin_amdgcn_wmma_f32_16x16x32_f16(false, a1, false, b1,
                                                 (short)0, acc[1][1], false, false);
        half16 b2 = ldfrag(bp2 + kk);
        acc[0][2] = __builtin_amdgcn_wmma_f32_16x16x32_f16(false, a0, false, b2,
                                                 (short)0, acc[0][2], false, false);
        acc[1][2] = __builtin_amdgcn_wmma_f32_16x16x32_f16(false, a1, false, b2,
                                                 (short)0, acc[1][2], false, false);
        half16 b3 = ldfrag(bp3 + kk);
        acc[0][3] = __builtin_amdgcn_wmma_f32_16x16x32_f16(false, a0, false, b3,
                                                 (short)0, acc[0][3], false, false);
        acc[1][3] = __builtin_amdgcn_wmma_f32_16x16x32_f16(false, a1, false, b3,
                                                 (short)0, acc[1][3], false, false);
    }

    // C/D layout: lane l holds col n = nt + (l&15); VGPR r holds row
    // m = tmg + (l>=16 ? 8 : 0) + r == tmg + khi + r.
    #pragma unroll
    for (int g = 0; g < 2; ++g) {
        const int mbase = tm + g * 16 + khi;
        #pragma unroll
        for (int t = 0; t < 4; ++t) {
            const int n = nbase + t * 16 + ln;
            if (mode == 0) {
                const float th = theta[n & (HEAD_DIM - 1)];
                #pragma unroll
                for (int r = 0; r < 8; ++r)
                    outH[(size_t)(mbase + r) * EMBED + n] =
                        (_Float16)cosf(acc[g][t][r] + th);
            } else {
                #pragma unroll
                for (int r = 0; r < 8; ++r)
                    outF[(size_t)(mbase + r) * EMBED + n] = acc[g][t][r];
            }
        }
    }
}

// ---------------------------------------------------------------------------
// Per-token attention over the HEADS axis (16x16 scores, softmax, 16x64 out).
// <1% of total FLOPs -> VALU + LDS; one 256-thread block per token.
// ---------------------------------------------------------------------------
__global__ __launch_bounds__(256)
void attn_heads(const _Float16* __restrict__ Q,
                const _Float16* __restrict__ K,
                const _Float16* __restrict__ V,
                _Float16* __restrict__ O) {
    __shared__ _Float16 sQ[HEADS * HEAD_DIM];
    __shared__ _Float16 sK[HEADS * HEAD_DIM];
    __shared__ _Float16 sV[HEADS * HEAD_DIM];
    __shared__ float    sS[HEADS * HEADS];

    const size_t base = (size_t)blockIdx.x * EMBED;
    const int tid = threadIdx.x;

    for (int i = tid; i < EMBED; i += 256) {
        sQ[i] = Q[base + i];
        sK[i] = K[base + i];
        sV[i] = V[base + i];
    }
    __syncthreads();

    // scores[i][j] = <q_i, k_j> / sqrt(64); one thread per (i,j)
    {
        const int i = tid >> 4, j = tid & 15;
        const _Float16* qi = &sQ[i * HEAD_DIM];
        const _Float16* kj = &sK[j * HEAD_DIM];
        float s = 0.f;
        #pragma unroll
        for (int d = 0; d < HEAD_DIM; ++d)
            s += (float)qi[d] * (float)kj[d];
        sS[i * HEADS + j] = s * 0.125f;
    }
    __syncthreads();

    // row softmax: thread i handles row i
    if (tid < HEADS) {
        float* row = &sS[tid * HEADS];
        float m = row[0];
        #pragma unroll
        for (int j = 1; j < HEADS; ++j) m = fmaxf(m, row[j]);
        float sum = 0.f;
        #pragma unroll
        for (int j = 0; j < HEADS; ++j) {
            float e = __expf(row[j] - m);
            row[j] = e;
            sum += e;
        }
        const float inv = 1.f / sum;
        #pragma unroll
        for (int j = 0; j < HEADS; ++j) row[j] *= inv;
    }
    __syncthreads();

    // out[i][d] = sum_j attn[i][j] * v[j][d]; 4 outputs per thread
    #pragma unroll
    for (int u = 0; u < 4; ++u) {
        const int idx = tid * 4 + u;          // 0..1023 == i*64 + d
        const int i = idx >> 6, d = idx & 63;
        const float* arow = &sS[i * HEADS];
        float acc = 0.f;
        #pragma unroll
        for (int j = 0; j < HEADS; ++j)
            acc += arow[j] * (float)sV[j * HEAD_DIM + d];
        O[base + idx] = (_Float16)acc;
    }
}

// ---------------------------------------------------------------------------
// Host launch
// ---------------------------------------------------------------------------
extern "C" void kernel_launch(void* const* d_in, const int* in_sizes, int n_in,
                              void* d_out, int out_size, void* d_ws, size_t ws_size,
                              hipStream_t stream) {
    const float* x     = (const float*)d_in[0];
    const float* Wk    = (const float*)d_in[1];
    const float* Wq    = (const float*)d_in[2];
    const float* Wv    = (const float*)d_in[3];
    const float* Wo    = (const float*)d_in[4];
    const float* theta = (const float*)d_in[5];

    // Workspace layout (f16 intermediates):
    //   [0, 64MB)        x16  -- reused as ao16 after QKV projections consume it
    //   [64MB, +8MB)     Wk16, Wq16, Wv16, Wo16 (2MB each)
    //   [72MB, +192MB)   qq, kq, vq (64MB each)
    char* ws = (char*)d_ws;
    const size_t XBYTES = (size_t)MROWS * EMBED * sizeof(_Float16);   // 64 MB
    const size_t WELEMS = (size_t)EMBED * EMBED;
    _Float16* x16  = (_Float16*)ws;
    _Float16* wk16 = (_Float16*)(ws + XBYTES);
    _Float16* wq16 = wk16 + WELEMS;
    _Float16* wv16 = wq16 + WELEMS;
    _Float16* wo16 = wv16 + WELEMS;
    _Float16* qq   = wo16 + WELEMS;
    _Float16* kq   = qq + (size_t)MROWS * EMBED;
    _Float16* vq   = kq + (size_t)MROWS * EMBED;
    _Float16* ao16 = x16;   // alias: x16 fully consumed before attn writes here

    // 1) convert inputs to f16
    const int NX = MROWS * EMBED;   // 33,554,432
    const int NW = EMBED * EMBED;   // 1,048,576
    cvt_f32_to_f16<<<(NX + 255) / 256, 256, 0, stream>>>(x,  x16,  NX);
    cvt_f32_to_f16<<<(NW + 255) / 256, 256, 0, stream>>>(Wk, wk16, NW);
    cvt_f32_to_f16<<<(NW + 255) / 256, 256, 0, stream>>>(Wq, wq16, NW);
    cvt_f32_to_f16<<<(NW + 255) / 256, 256, 0, stream>>>(Wv, wv16, NW);
    cvt_f32_to_f16<<<(NW + 255) / 256, 256, 0, stream>>>(Wo, wo16, NW);

    // 2) QKV projections + fused cos(. + theta) epilogue (WMMA)
    dim3 gblk(32, 4);
    dim3 ggrid(MROWS / 32, EMBED / 256);
    gemm_wmma<<<ggrid, gblk, 0, stream>>>(x16, wk16, kq, nullptr, theta, 0);
    gemm_wmma<<<ggrid, gblk, 0, stream>>>(x16, wq16, qq, nullptr, theta, 0);
    gemm_wmma<<<ggrid, gblk, 0, stream>>>(x16, wv16, vq, nullptr, theta, 0);

    // 3) per-token head attention (VALU)
    attn_heads<<<MROWS, 256, 0, stream>>>(qq, kq, vq, ao16);

    // 4) final projection out = ao @ Wo^T, f32 output (WMMA)
    gemm_wmma<<<ggrid, gblk, 0, stream>>>(ao16, wo16, nullptr, (float*)d_out,
                                          nullptr, 1);
}
